// SharpReconLoss_63385127354985
// MI455X (gfx1250) — matrix operand order repaired and checked by reading, not compiled, not run on hardware
//
#include <hip/hip_runtime.h>

// ---------------------------------------------------------------------------
// SharpReconLoss for MI455X (gfx1250, wave32, WMMA + async global->LDS).
// 11x11 box filters (the compute-dominant part of SSIM) run on the matrix
// pipes via v_wmma_f32_16x16x32_f16 with constant banded matrices; cheap 3x3
// stencils + L1 stay on VALU. One wave owns one 16x16 output tile. Interior
// tiles stage their 32x32 halo via GLOBAL_LOAD_ASYNC_TO_LDS (ASYNCcnt path),
// boundary tiles use guarded zero-padded loads.
// ---------------------------------------------------------------------------

typedef __attribute__((ext_vector_type(16))) _Float16 v16h;
typedef __attribute__((ext_vector_type(8)))  float    v8f;

typedef __attribute__((address_space(1))) int gint_t;   // global (as expected by builtin)
typedef __attribute__((address_space(3))) int sint_t;   // LDS

#define H_IMG 512
#define W_IMG 512
#define TILE  16
#define HALO  5
#define LT    32            // LDS tile dim (16 + 2*5 halo, padded to 32)
#define WPB   4             // waves per block (block = 128 threads)
#define NPIX  16777216.0f   // 64*512*512

#if defined(__gfx1250__) && __has_builtin(__builtin_amdgcn_global_load_async_to_lds_b32)
#define USE_ASYNC_LDS 1
#else
#define USE_ASYNC_LDS 0
#endif

__device__ __forceinline__ float wave_reduce32(float x) {
  #pragma unroll
  for (int off = 16; off > 0; off >>= 1) x += __shfl_down(x, off, 32);
  return x;
}

__global__ void __launch_bounds__(32) srl_init(float* acc) {
  if (threadIdx.x < 8) acc[threadIdx.x] = 0.0f;
}

__global__ void __launch_bounds__(128)
srl_main(const float* __restrict__ pred, const float* __restrict__ targ,
         float* __restrict__ acc) {
  __shared__ float sP[WPB][LT][LT];
  __shared__ float sT[WPB][LT][LT];
  __shared__ float sH[WPB][32][16];      // horizontal box sums (f32)
  __shared__ float sRed[WPB][4];

  const int lane = threadIdx.x & 31;
  const int wid  = threadIdx.x >> 5;
  const int gw   = blockIdx.x * WPB + wid;      // 65536 wave-tiles total
  const int img  = gw >> 10;                    // 1024 tiles (32x32) per image
  const int t    = gw & 1023;
  const int y0   = (t >> 5) * TILE;
  const int x0   = (t & 31) * TILE;
  const size_t base = (size_t)img * (H_IMG * W_IMG);

  const int hi = lane >> 4;                     // half-wave selector
  const int mn = lane & 15;                     // m or n coordinate

  // -------- stage 32x32 halo tiles of pred/target into LDS ---------------
  {
    size_t pf = base + (size_t)min(max(y0 - HALO, 0), H_IMG - 1) * W_IMG +
                (size_t)min(max(x0 - HALO, 0), W_IMG - 1);
    __builtin_prefetch(pred + pf, 0, 1);        // global_prefetch_b8
    __builtin_prefetch(targ + pf, 0, 1);
  }
  // wave-uniform: does the 32x32 halo stay fully inside the image?
  const bool interior = (x0 >= HALO) && (y0 >= HALO) &&
                        (x0 - HALO + LT <= W_IMG) && (y0 - HALO + LT <= H_IMG);
#if USE_ASYNC_LDS
  if (interior) {
    // async copy engine: global -> LDS directly, no VGPR staging (ASYNCcnt)
    const float* gp = pred + base + (size_t)(y0 - HALO) * W_IMG + (x0 - HALO) + lane;
    const float* gt = targ + base + (size_t)(y0 - HALO) * W_IMG + (x0 - HALO) + lane;
    #pragma unroll 8
    for (int r = 0; r < LT; ++r) {
      __builtin_amdgcn_global_load_async_to_lds_b32(
          (gint_t*)(gp + (size_t)r * W_IMG), (sint_t*)&sP[wid][r][lane], 0, 0);
      __builtin_amdgcn_global_load_async_to_lds_b32(
          (gint_t*)(gt + (size_t)r * W_IMG), (sint_t*)&sT[wid][r][lane], 0, 0);
    }
#if __has_builtin(__builtin_amdgcn_s_wait_asynccnt)
    __builtin_amdgcn_s_wait_asynccnt(0);
#else
    asm volatile("s_wait_asynccnt 0" ::: "memory");
#endif
  } else
#endif
  {
    const int gx  = x0 - HALO + lane;
    const bool xin = (gx >= 0) && (gx < W_IMG);
    #pragma unroll 4
    for (int r = 0; r < LT; ++r) {
      const int gy = y0 - HALO + r;
      const bool in = xin && (gy >= 0) && (gy < H_IMG);
      const size_t idx = base + (size_t)gy * W_IMG + gx;
      sP[wid][r][lane] = in ? pred[idx] : 0.0f;
      sT[wid][r][lane] = in ? targ[idx] : 0.0f;
    }
  }
  __syncthreads();

  // -------- constant banded matrices (f16) -------------------------------
  // Horizontal B (32x16): B[k][n] = 1 for n <= k <= n+10.
  // B k order used here (contiguous): lane -> n = lane&15, K pair per VGPR v = {2v+16*hi, +1}.
  v16h Bh = {};
  #pragma unroll
  for (int v = 0; v < 8; ++v) {
    const int k0 = 2 * v + 16 * hi;
    Bh[2 * v]     = (_Float16)(((k0     >= mn) && (k0     <= mn + 10)) ? 1.0f : 0.0f);
    Bh[2 * v + 1] = (_Float16)(((k0 + 1 >= mn) && (k0 + 1 <= mn + 10)) ? 1.0f : 0.0f);
  }
  // Vertical A (16x32): A2[m][k] = 1 for m <= k <= m+10 (H rows >=26 masked).
  // A k order used here (contiguous): lane -> m = lane&15, K pair per VGPR v = {2v+16*hi, +1}.
  v16h Av = {};
  #pragma unroll
  for (int v = 0; v < 8; ++v) {
    const int k0 = 2 * v + 16 * hi;
    Av[2 * v]     = (_Float16)(((k0     >= mn) && (k0     <= mn + 10)) ? 1.0f : 0.0f);
    Av[2 * v + 1] = (_Float16)(((k0 + 1 >= mn) && (k0 + 1 <= mn + 10)) ? 1.0f : 0.0f);
  }

  const float (*P)[LT] = sP[wid];
  const float (*T)[LT] = sT[wid];

  // -------- 5 fields: p, t, p*p, t*t, p*t -> 11x11 box sums via WMMA -----
  v8f box[5];
  #pragma unroll
  for (int f = 0; f < 5; ++f) {
    // horizontal pass: two 16-row halves of the 26-row halo
    #pragma unroll
    for (int half = 0; half < 2; ++half) {
      v16h A = {};
      const int row = 16 * half + mn;
      #pragma unroll
      for (int v = 0; v < 8; ++v) {
        const int k0 = 2 * v + 16 * hi;
        #pragma unroll
        for (int e = 0; e < 2; ++e) {
          const int k = k0 + e;
          const float p = P[row][k], tt = T[row][k];
          float val;
          if      (f == 0) val = p;
          else if (f == 1) val = tt;
          else if (f == 2) val = p * p;
          else if (f == 3) val = tt * tt;
          else             val = p * tt;
          A[2 * v + e] = (_Float16)val;
        }
      }
      v8f c = {};
      c = __builtin_amdgcn_wmma_f32_16x16x32_f16(false, A, false, Bh,
                                                 (short)0, c, false, false);
      // C layout: element v lives at (M = v + 8*hi, N = mn) -> H row 16*half+M
      #pragma unroll
      for (int v = 0; v < 8; ++v)
        sH[wid][16 * half + v + 8 * hi][mn] = c[v];
    }
    __syncthreads();

    // vertical pass: B2 = H (32x16, rows >= 26 zeroed by the band in Av)
    v16h B2 = {};
    #pragma unroll
    for (int v = 0; v < 8; ++v) {
      const int k0 = 2 * v + 16 * hi;
      B2[2 * v]     = (_Float16)sH[wid][k0][mn];
      B2[2 * v + 1] = (_Float16)sH[wid][k0 + 1][mn];
    }
    v8f c2 = {};
    c2 = __builtin_amdgcn_wmma_f32_16x16x32_f16(false, Av, false, B2,
                                                (short)0, c2, false, false);
    box[f] = c2;
    __syncthreads();   // before next field overwrites sH
  }

  // -------- epilogue: SSIM map + L1 + Laplacian + Sobel on VALU ----------
  const float inv121 = 1.0f / 121.0f;
  const float C1 = 1e-4f, C2 = 9e-4f;
  float s_l1 = 0.f, s_ssim = 0.f, s_edge = 0.f, s_gxy = 0.f;
  #pragma unroll
  for (int i = 0; i < 8; ++i) {
    const float mu_p = box[0][i] * inv121;
    const float mu_t = box[1][i] * inv121;
    const float spp  = box[2][i] * inv121 - mu_p * mu_p;
    const float stt  = box[3][i] * inv121 - mu_t * mu_t;
    const float spt  = box[4][i] * inv121 - mu_p * mu_t;
    s_ssim += ((2.f * mu_p * mu_t + C1) * (2.f * spt + C2)) /
              ((mu_p * mu_p + mu_t * mu_t + C1) * (spp + stt + C2));

    const int r = i + 8 * hi + HALO;   // this lane's output pixel in the tile
    const int c = mn + HALO;
    float d[3][3];
    #pragma unroll
    for (int dy = 0; dy < 3; ++dy)
      #pragma unroll
      for (int dx = 0; dx < 3; ++dx)
        d[dy][dx] = P[r - 1 + dy][c - 1 + dx] - T[r - 1 + dy][c - 1 + dx];

    s_l1 += fabsf(d[1][1]);
    s_edge += fabsf(d[0][1] + d[2][1] + d[1][0] + d[1][2] - 4.f * d[1][1]);
    const float gx = -d[0][0] + d[0][2] - 2.f * d[1][0] + 2.f * d[1][2] - d[2][0] + d[2][2];
    const float gy = -d[0][0] - 2.f * d[0][1] - d[0][2] + d[2][0] + 2.f * d[2][1] + d[2][2];
    s_gxy += fabsf(gx) + fabsf(gy);
  }

  // -------- reduce: wave -> block -> global atomics ----------------------
  float part[4] = { s_l1, s_ssim, s_edge, s_gxy };
  #pragma unroll
  for (int j = 0; j < 4; ++j) part[j] = wave_reduce32(part[j]);
  if (lane == 0) {
    #pragma unroll
    for (int j = 0; j < 4; ++j) sRed[wid][j] = part[j];
  }
  __syncthreads();
  if (threadIdx.x == 0) {
    float sum[4] = {0.f, 0.f, 0.f, 0.f};
    for (int w = 0; w < WPB; ++w)
      for (int j = 0; j < 4; ++j) sum[j] += sRed[w][j];
    for (int j = 0; j < 4; ++j) atomicAdd(&acc[j], sum[j]);
  }
}

__global__ void __launch_bounds__(32) srl_fin(const float* acc, float* out) {
  if (threadIdx.x != 0) return;
  const float l1   = acc[0] / NPIX;
  const float ssim = 1.0f - acc[1] / NPIX;
  const float edge = acc[2] / NPIX;
  const float grad = acc[3] / NPIX;   // |sobel_x| mean + |sobel_y| mean
  out[0] = l1 + 0.5f * ssim + 0.3f * edge + 0.2f * grad;
  out[1] = l1;
  out[2] = ssim;
  out[3] = edge;
  out[4] = grad;
}

extern "C" void kernel_launch(void* const* d_in, const int* in_sizes, int n_in,
                              void* d_out, int out_size, void* d_ws, size_t ws_size,
                              hipStream_t stream) {
  const float* pred = (const float*)d_in[0];
  const float* targ = (const float*)d_in[1];
  float* out = (float*)d_out;
  float* acc = (float*)d_ws;

  srl_init<<<1, 32, 0, stream>>>(acc);
  // 64 images * (32x32) tiles = 65536 wave-tiles; 4 waves/block -> 16384 blocks
  srl_main<<<16384, 128, 0, stream>>>(pred, targ, acc);
  srl_fin<<<1, 32, 0, stream>>>(acc, out);
}
